// RandomFeature_85641647882473
// MI455X (gfx1250) — compile-verified, matrix operand order checked by reading, not run on hardware
//
#include <hip/hip_runtime.h>
#include <math.h>

typedef __attribute__((ext_vector_type(16))) _Float16 v16h;
typedef __attribute__((ext_vector_type(8)))  float    v8f;

#define RS_INV 4.0f  // 1 / rs, rs = 0.25

#if __has_builtin(__builtin_amdgcn_tanhf)
#define FAST_TANH(x) __builtin_amdgcn_tanhf(x)
#else
#define FAST_TANH(x) tanhf(x)
#endif

static __device__ __forceinline__ unsigned wave_ballot32(bool pred) {
#if __has_builtin(__builtin_amdgcn_ballot_w32)
    return __builtin_amdgcn_ballot_w32(pred);
#else
    return (unsigned)__ballot(pred);
#endif
}

// One wave32 handles one 16-point tile of n.
// WMMA tile: M = 16 points, N = 16 features (j), K = 32 (only K=0,1 carry D=2).
__global__ __launch_bounds__(256)
void RandomFeature_85641647882473_kernel(const float* __restrict__ p,
                                         const float* __restrict__ cs,
                                         const float* __restrict__ W,
                                         const float* __restrict__ b,
                                         const float* __restrict__ um,
                                         float* __restrict__ out,
                                         int N, int Mp, int Jn)
{
    const int lane = threadIdx.x & 31;
    const int wave = threadIdx.x >> 5;
    const int tile = blockIdx.x * (blockDim.x >> 5) + wave;
    const int n0   = tile * 16;

    const int  sub = lane & 15;          // column / point index within half-wave
    const bool lo  = lane < 16;          // low half-wave holds the live K=0,1 slots
    const int  hi8 = (lane >> 4) << 3;   // row offset of this half-wave in C/D tile

    // Point coordinates for this lane's A-row (lanes >= 16 duplicate; unused there).
    const float2 pt = ((const float2*)p)[n0 + sub];

    float acc[8];
#pragma unroll
    for (int r = 0; r < 8; ++r) acc[r] = 0.0f;

    for (int m = 0; m < Mp; ++m) {
        const float2 c  = ((const float2*)cs)[m];
        const float  sx = (pt.x - c.x) * RS_INV;
        const float  sy = (pt.y - c.y) * RS_INV;

        // Partition-of-unity indicator, one bit per point (lanes 0..15 only).
        const bool inside = lo && (fabsf(sx) <= 1.0f) && (fabsf(sy) <= 1.0f);
        const unsigned mb = wave_ballot32(inside);   // wave-uniform scalar
        if (mb == 0u) continue;                      // uniform skip: EXEC stays full

        // A tile (16x32 f16): K=0,1 = std, rest zero. Lanes>=16 hold K=8,9 -> zero.
        v16h a = {};
        a[0] = lo ? (_Float16)sx : (_Float16)0.0f;
        a[1] = lo ? (_Float16)sy : (_Float16)0.0f;

        // Per-row mask as floats (row M = r + hi8).
        float mf[8];
#pragma unroll
        for (int r = 0; r < 8; ++r)
            mf[r] = (float)((mb >> (r + hi8)) & 1u);

        const float* __restrict__ Wm  = W  + m * (2 * Jn);
        const float* __restrict__ bm  = b  + m * Jn;
        const float* __restrict__ umm = um + m * Jn;

        for (int jc = 0; jc < Jn; jc += 16) {
            const int j = jc + sub;

            // B tile (32x16 f16): K=0,1 = W[m, 0/1, j], rest zero.
            const float w0 = Wm[j];        // d = 0
            const float w1 = Wm[Jn + j];   // d = 1
            v16h bt = {};
            bt[0] = lo ? (_Float16)w0 : (_Float16)0.0f;
            bt[1] = lo ? (_Float16)w1 : (_Float16)0.0f;

            // C = bias broadcast down the column (rows share b[m, j]).
            const float bias = bm[j];
            v8f cacc;
#pragma unroll
            for (int r = 0; r < 8; ++r) cacc[r] = bias;

            // D = A*B + C  ->  pre-activation h for 16 points x 16 features
            cacc = __builtin_amdgcn_wmma_f32_16x16x32_f16(
                false, a, false, bt, (short)0, cacc, false, false);

            const float u = umm[j];
#pragma unroll
            for (int r = 0; r < 8; ++r)
                acc[r] += FAST_TANH(cacc[r]) * u * mf[r];
        }
    }

    // Reduce over j (lane dimension) within each 16-lane half-wave.
#pragma unroll
    for (int r = 0; r < 8; ++r) {
        float v = acc[r];
        v += __shfl_xor(v, 1, 32);
        v += __shfl_xor(v, 2, 32);
        v += __shfl_xor(v, 4, 32);
        v += __shfl_xor(v, 8, 32);
        acc[r] = v;
    }

    // Lane 0 writes points n0+0..7, lane 16 writes n0+8..15.
    if (sub == 0) {
#pragma unroll
        for (int r = 0; r < 8; ++r)
            out[n0 + hi8 + r] = acc[r];
    }
}

extern "C" void kernel_launch(void* const* d_in, const int* in_sizes, int n_in,
                              void* d_out, int out_size, void* d_ws, size_t ws_size,
                              hipStream_t stream) {
    const float* p  = (const float*)d_in[0];   // [N, 2]
    const float* cs = (const float*)d_in[1];   // [Mp, 2]
    const float* W  = (const float*)d_in[2];   // [Mp, 2, Jn]
    const float* b  = (const float*)d_in[3];   // [Mp, Jn]
    const float* um = (const float*)d_in[4];   // [Mp, Jn]
    float* out = (float*)d_out;                // [N, 1]

    const int N  = in_sizes[0] / 2;
    const int Mp = in_sizes[1] / 2;
    const int Jn = (Mp > 0) ? (in_sizes[3] / Mp) : 0;

    const int threads       = 256;           // 8 waves per block
    const int wavesPerBlock = threads / 32;
    const int tiles         = N / 16;        // 2048 tiles for N=32768
    const int blocks        = tiles / wavesPerBlock;

    RandomFeature_85641647882473_kernel<<<blocks, threads, 0, stream>>>(
        p, cs, W, b, um, out, N, Mp, Jn);
}
